// CGCNNConvSimple_74637941670346
// MI455X (gfx1250) — matrix operand order repaired and checked by reading, not compiled, not run on hardware
//
#include <hip/hip_runtime.h>

// ---------------------------------------------------------------------------
// CGCNN conv, fused, MI455X (gfx1250): bf16 WMMA, LDS-resident weights,
// 32-edge M-tiles (2x B-fragment reuse), XOR-swizzled activation staging.
// ---------------------------------------------------------------------------

typedef __attribute__((ext_vector_type(16))) __bf16 bf16x16;
typedef __attribute__((ext_vector_type(8)))  __bf16 bf16x8;
typedef __attribute__((ext_vector_type(8)))  float  f32x8;

#define NODE_DIM 128
#define N_NODES  10000
#define N_EDGES  640000
#define TILE_M   32
#define N_TILES  (N_EDGES / TILE_M)  // 20000 tiles of 32 edges
#define ASTRIDE  384                 // unpadded; bank conflicts avoided via XOR swizzle
#define WFRAGS   224                 // 96 (eW1) + 32 (eW2) + 64 (nW1) + 32 (nW2)
#define WELEMS   (WFRAGS * 512)      // bf16 elements of formatted weights (229376 B)

__device__ __forceinline__ unsigned int f2bf1(float f) {
  unsigned int u = __float_as_uint(f);
  return (u + 0x7fffu + ((u >> 16) & 1u)) >> 16;   // RNE f32 -> bf16
}
__device__ __forceinline__ unsigned int pk2bf(float lo, float hi) {
#if __has_builtin(__builtin_amdgcn_cvt_pk_bf16_f32)
  auto v = __builtin_amdgcn_cvt_pk_bf16_f32(lo, hi);
  return __builtin_bit_cast(unsigned int, v);
#else
  return f2bf1(lo) | (f2bf1(hi) << 16);
#endif
}
__device__ __forceinline__ unsigned short f2bf_s(float f) {   // single-element convert
#if __has_builtin(__builtin_amdgcn_cvt_pk_bf16_f32)
  return (unsigned short)(pk2bf(f, f) & 0xffffu);             // one v_cvt_pk_bf16_f32
#else
  return (unsigned short)f2bf1(f);
#endif
}
__device__ __forceinline__ float softplusf(float x) {
  return x > 15.f ? x : __logf(1.f + __expf(x));
}
// swizzled element index into the [32 x 384] staging buffer: XOR on 16-elem
// granules (col bits 4-6) by row -> 16 B alignment preserved for b128 DS ops
__device__ __forceinline__ int aidx(int row, int col) {
  return row * ASTRIDE + (col ^ ((row & 7) << 4));
}

// ---------------------------------------------------------------------------
// Prep: f32 weights [fan_in, 128] row-major -> bf16 WMMA B-fragment layout.
// 16-bit B 32x16 (this code's contiguous k, same as A reads): lanes 0-15 K=kb*32+e, lanes 16-31 K=kb*32+16+e,
// column = nb*16 + lane%16.  Fragment order: eW1[kb][nb], eW2, nW1, nW2.
// ---------------------------------------------------------------------------
__global__ void fmt_weights(const float* __restrict__ nW1, const float* __restrict__ nW2,
                            const float* __restrict__ eW1, const float* __restrict__ eW2,
                            unsigned short* __restrict__ wout) {
  int t = blockIdx.x * 256 + threadIdx.x;
  if (t >= WELEMS) return;
  int frag = t >> 9;
  int r    = t & 511;
  int lane = r >> 4;
  int e    = r & 15;
  const float* W; int fl;
  if      (frag <  96) { W = eW1; fl = frag;       }
  else if (frag < 128) { W = eW2; fl = frag - 96;  }
  else if (frag < 192) { W = nW1; fl = frag - 128; }
  else                 { W = nW2; fl = frag - 192; }
  int kb = fl >> 3, nb = fl & 7;
  int k = kb * 32 + ((lane >> 4) << 4) + e;
  int n = nb * 16 + (lane & 15);
  wout[t] = (unsigned short)f2bf1(W[k * 128 + n]);
}

__global__ void zero_xnew(float* __restrict__ p) {
  int t = blockIdx.x * 256 + threadIdx.x;
  if (t < N_NODES * NODE_DIM) p[t] = 0.f;
}

#define WMMA_BF16(c, a, b) \
  c = __builtin_amdgcn_wmma_f32_16x16x32_bf16(false, a, false, b, (short)0, c, false, false)

// ---------------------------------------------------------------------------
// 32x128 GEMM: two 16-row M-tiles share each B fragment (2x LDS B reuse).
// B fragments loaded in batches of 4 ahead of 8 WMMAs -> partial dscnt waits.
// ---------------------------------------------------------------------------
template <int KB, int SPLIT, int C0, int C1>
__device__ __forceinline__ void gemm32(const unsigned short* __restrict__ A,
                                       const unsigned short* __restrict__ wlds,
                                       int fragBase, int lane,
                                       f32x8 accA[8], f32x8 accB[8]) {
  const int r  = lane & 15;
  const int ho = (lane >> 4) * 16;
  const int sw = (r & 7) << 4;                      // (r+16)&7 == r&7: same swizzle
  const unsigned short* rowA  = A + r * ASTRIDE;
  const unsigned short* rowB  = A + (r + 16) * ASTRIDE;
  const unsigned short* bbase = wlds + fragBase * 512 + lane * 16;
#pragma unroll
  for (int kb = 0; kb < KB; ++kb) {
    const int col = (kb < SPLIT) ? (C0 + kb * 32) : (C1 + (kb - SPLIT) * 32);
    const int q0 = (col + ho) ^ sw;
    const int q1 = (col + ho + 8) ^ sw;
    bf16x8 a0 = *(const bf16x8*)(rowA + q0);
    bf16x8 a1 = *(const bf16x8*)(rowA + q1);
    bf16x8 b0 = *(const bf16x8*)(rowB + q0);
    bf16x8 b1 = *(const bf16x8*)(rowB + q1);
    bf16x16 av0 = __builtin_shufflevector(a0, a1, 0,1,2,3,4,5,6,7,8,9,10,11,12,13,14,15);
    bf16x16 av1 = __builtin_shufflevector(b0, b1, 0,1,2,3,4,5,6,7,8,9,10,11,12,13,14,15);
    const unsigned short* bp = bbase + kb * (8 * 512);
    bf16x16 bv0 = *(const bf16x16*)(bp + 0 * 512);
    bf16x16 bv1 = *(const bf16x16*)(bp + 1 * 512);
    bf16x16 bv2 = *(const bf16x16*)(bp + 2 * 512);
    bf16x16 bv3 = *(const bf16x16*)(bp + 3 * 512);
    WMMA_BF16(accA[0], av0, bv0);  WMMA_BF16(accB[0], av1, bv0);
    WMMA_BF16(accA[1], av0, bv1);  WMMA_BF16(accB[1], av1, bv1);
    WMMA_BF16(accA[2], av0, bv2);  WMMA_BF16(accB[2], av1, bv2);
    WMMA_BF16(accA[3], av0, bv3);  WMMA_BF16(accB[3], av1, bv3);
    bv0 = *(const bf16x16*)(bp + 4 * 512);
    bv1 = *(const bf16x16*)(bp + 5 * 512);
    bv2 = *(const bf16x16*)(bp + 6 * 512);
    bv3 = *(const bf16x16*)(bp + 7 * 512);
    WMMA_BF16(accA[4], av0, bv0);  WMMA_BF16(accB[4], av1, bv0);
    WMMA_BF16(accA[5], av0, bv1);  WMMA_BF16(accB[5], av1, bv1);
    WMMA_BF16(accA[6], av0, bv2);  WMMA_BF16(accB[6], av1, bv2);
    WMMA_BF16(accA[7], av0, bv3);  WMMA_BF16(accB[7], av1, bv3);
  }
}

// bias + softplus + bf16 store into staging columns [colBase, colBase+128)
// C layout: lanes 0-15 -> N=lane, M=v ; lanes 16-31 -> N=lane-16, M=v+8
__device__ __forceinline__ void store_act32(const f32x8 cA[8], const f32x8 cB[8],
                                            const float* __restrict__ bias,
                                            unsigned short* __restrict__ A,
                                            int colBase, int lane) {
  const int N  = lane & 15;
  const int mh = (lane >> 4) * 8;
#pragma unroll
  for (int nb = 0; nb < 8; ++nb) {
    float b   = bias[nb * 16 + N];
    int   col = colBase + nb * 16 + N;
#pragma unroll
    for (int v = 0; v < 8; ++v) {
      int M = v + mh;
      A[aidx(M, col)]      = f2bf_s(softplusf(cA[nb][v] + b));
      A[aidx(M + 16, col)] = f2bf_s(softplusf(cB[nb][v] + b));
    }
  }
}

__global__ __launch_bounds__(128) void gnn_fused(
    const float* __restrict__ x,  const int* __restrict__ ei,
    const float* __restrict__ ea,
    const float* __restrict__ pnb1, const float* __restrict__ pnb2,
    const float* __restrict__ peb1, const float* __restrict__ peb2,
    const unsigned short* __restrict__ wfmt,
    float* __restrict__ out) {
  extern __shared__ unsigned short smem[];
  unsigned short* wlds = smem;                                 // 229376 B weights
  const int lane = threadIdx.x & 31;
  const int wave = threadIdx.x >> 5;
  unsigned short* A = smem + WELEMS + wave * (TILE_M * ASTRIDE);  // 24576 B / wave

  // cooperative weight load: bf16 fragments global(L2) -> LDS
  {
    const uint4* src = (const uint4*)wfmt;
    uint4*       dst = (uint4*)smem;
#pragma unroll 4
    for (int i = threadIdx.x; i < WELEMS / 8; i += 128) dst[i] = src[i];
  }
  __syncthreads();

  float* xnew  = out;
  float* eanew = out + N_NODES * NODE_DIM;

  const int slot   = blockIdx.x * 4 + wave;
  const int nslots = gridDim.x * 4;
  const f32x8 z8 = {0.f, 0.f, 0.f, 0.f, 0.f, 0.f, 0.f, 0.f};

  for (int tile = slot; tile < N_TILES; tile += nslots) {
    const int ebase = tile * TILE_M;

    // per-lane edge indices: lane l holds row[ebase+l] and col[ebase+l]
    int ivr = ei[ebase + lane];
    int ivc = ei[N_EDGES + ebase + lane];

    // prefetch next tile's edge_attr stream (16 KB) into cache
    {
      int nxt = ebase + nslots * TILE_M;
      if (nxt < N_EDGES) {
        const char* p = (const char*)(ea + (size_t)nxt * 128);
#pragma unroll
        for (int j = 0; j < 4; ++j)
          __builtin_prefetch(p + lane * 512 + j * 128, 0, 1);
      }
    }

    // ---- gather edge_input = [x_row | x_col | edge_attr] -> bf16 LDS -------
    const float4* x4  = (const float4*)x;
    const float4* ea4 = (const float4*)ea;
#pragma unroll
    for (int e = 0; e < TILE_M; ++e) {
      int re = __shfl(ivr, e, 32);                 // uniform -> readlane + saddr
      int ce = __shfl(ivc, e, 32);
      float4 fa = x4[re * 32 + lane];
      float4 fb = x4[ce * 32 + lane];
      float4 fc = ea4[(ebase + e) * 32 + lane];
      uint2 pa, pb, pc;
      pa.x = pk2bf(fa.x, fa.y);  pa.y = pk2bf(fa.z, fa.w);
      pb.x = pk2bf(fb.x, fb.y);  pb.y = pk2bf(fb.z, fb.w);
      pc.x = pk2bf(fc.x, fc.y);  pc.y = pk2bf(fc.z, fc.w);
      *(uint2*)(A + aidx(e, lane * 4))       = pa;  // cols   0..127  x_row
      *(uint2*)(A + aidx(e, 128 + lane * 4)) = pb;  // cols 128..255  x_col
      *(uint2*)(A + aidx(e, 256 + lane * 4)) = pc;  // cols 256..383  edge_attr
    }

    f32x8 accA[8], accB[8];

    // ---- GEMM1: edge_input[32x384] @ eW1 -> softplus -> h (cols 128..) -----
#pragma unroll
    for (int nb = 0; nb < 8; ++nb) { accA[nb] = z8; accB[nb] = z8; }
    gemm32<12, 12, 0, 0>(A, wlds, 0, lane, accA, accB);
    store_act32(accA, accB, peb1, A, 128, lane);      // overwrites dead x_col

    // ---- GEMM2: h[32x128] @ eW2 -> edge_attr_new (global f32 + LDS 256..) --
#pragma unroll
    for (int nb = 0; nb < 8; ++nb) { accA[nb] = z8; accB[nb] = z8; }
    gemm32<4, 4, 128, 0>(A, wlds, 96, lane, accA, accB);
    {
      const int N = lane & 15, mh = (lane >> 4) * 8;
      // one base pointer per M-tile; all stores use compile-time imm offsets
      float* ebp0 = eanew + (size_t)(unsigned)((ebase + mh) * 128 + N);
      float* ebp1 = ebp0 + 16 * 128;
#pragma unroll
      for (int nb = 0; nb < 8; ++nb) {
        float b = peb2[nb * 16 + N];
#pragma unroll
        for (int v = 0; v < 8; ++v) {
          float t0 = accA[nb][v] + b;
          float t1 = accB[nb][v] + b;
          ebp0[v * 128 + nb * 16] = t0;
          ebp1[v * 128 + nb * 16] = t1;
          int M0 = v + mh;
          A[aidx(M0, 256 + nb * 16 + N)]      = f2bf_s(t0);  // dead edge_attr cols
          A[aidx(M0 + 16, 256 + nb * 16 + N)] = f2bf_s(t1);
        }
      }
    }

    // ---- GEMM3: [x_row | edge_new][32x256] @ nW1 -> softplus -> m ----------
#pragma unroll
    for (int nb = 0; nb < 8; ++nb) { accA[nb] = z8; accB[nb] = z8; }
    gemm32<8, 4, 0, 256>(A, wlds, 128, lane, accA, accB);
    store_act32(accA, accB, pnb1, A, 128, lane);      // overwrites dead h

    // ---- GEMM4: m[32x128] @ nW2 -> messages -> scatter-add into x_new ------
#pragma unroll
    for (int nb = 0; nb < 8; ++nb) { accA[nb] = z8; accB[nb] = z8; }
    gemm32<4, 4, 128, 0>(A, wlds, 192, lane, accA, accB);
    {
      const int N = lane & 15, mh = (lane >> 4) * 8;
      float bb[8];
#pragma unroll
      for (int nb = 0; nb < 8; ++nb) bb[nb] = pnb2[nb * 16 + N];
#pragma unroll
      for (int v = 0; v < 8; ++v) {
        int M0 = v + mh;
        int d0 = __shfl(ivc, M0, 32);                 // destination node = col[M]
        int d1 = __shfl(ivc, M0 + 16, 32);
        // explicit zext -> saddr + 32-bit voffset addressing, imm offsets per nb
        float* p0 = xnew + (size_t)(unsigned)(d0 * 128 + N);
        float* p1 = xnew + (size_t)(unsigned)(d1 * 128 + N);
#pragma unroll
        for (int nb = 0; nb < 8; ++nb) {
          unsafeAtomicAdd(p0 + nb * 16, accA[nb][v] + bb[nb]);
          unsafeAtomicAdd(p1 + nb * 16, accB[nb][v] + bb[nb]);
        }
      }
    }
  }
}

// ---------------------------------------------------------------------------
extern "C" void kernel_launch(void* const* d_in, const int* in_sizes, int n_in,
                              void* d_out, int out_size, void* d_ws, size_t ws_size,
                              hipStream_t stream) {
  const float* x   = (const float*)d_in[0];
  const int*   ei  = (const int*)  d_in[1];
  const float* ea  = (const float*)d_in[2];
  const float* nW1 = (const float*)d_in[3];
  const float* nb1 = (const float*)d_in[4];
  const float* nW2 = (const float*)d_in[5];
  const float* nb2 = (const float*)d_in[6];
  const float* eW1 = (const float*)d_in[7];
  const float* eb1 = (const float*)d_in[8];
  const float* eW2 = (const float*)d_in[9];
  const float* eb2 = (const float*)d_in[10];
  unsigned short* wfmt = (unsigned short*)d_ws;   // 229376 B formatted bf16 weights
  float* out = (float*)d_out;

  fmt_weights<<<(WELEMS + 255) / 256, 256, 0, stream>>>(nW1, nW2, eW1, eW2, wfmt);
  zero_xnew<<<(N_NODES * NODE_DIM + 255) / 256, 256, 0, stream>>>(out);

  // 229376 (weights) + 4 * 24576 (staging) = 327680 B = exactly 320 KB/WGP
  size_t lds = (size_t)WELEMS * 2 + 4u * TILE_M * ASTRIDE * 2u;
  gnn_fused<<<dim3(1024), dim3(128), lds, stream>>>(x, ei, ea, nb1, nb2, eb1, eb2, wfmt, out);
}
